// Attention_5342939316549
// MI455X (gfx1250) — matrix operand order repaired and checked by reading, not compiled, not run on hardware
//
#include <hip/hip_runtime.h>
#include <hip/hip_bf16.h>

// ---------------------------------------------------------------------------
// Problem constants (from reference)
// ---------------------------------------------------------------------------
#define B_    4
#define S0_   768
#define S1_   256
#define T_    1024
#define D0_   2048
#define D1_   1024
#define NH_   8      // query heads
#define H_    256    // head dim
#define QKVC  2560   // N*H + 2*H  (q | k | v columns)
#define BIG_NEG (-2.3819763e+38f)

typedef unsigned short u16;
typedef __attribute__((ext_vector_type(16))) __bf16        bf16x16;
typedef __attribute__((ext_vector_type(8)))  float         f32x8;
typedef __attribute__((ext_vector_type(4)))  unsigned int  u32x4;

// ---------------------------------------------------------------------------
// helpers
// ---------------------------------------------------------------------------
__device__ inline u16 f2bf(float f) {
    union { float f; unsigned u; } a; a.f = f;
    unsigned r = a.u + 0x7fffu + ((a.u >> 16) & 1u);   // round-to-nearest-even
    return (u16)(r >> 16);
}

__device__ inline f32x8 fzero8() {
    f32x8 r;
    #pragma unroll
    for (int i = 0; i < 8; ++i) r[i] = 0.0f;
    return r;
}

// Load a 16-half fragment as two 16B pieces at p and p+gap (halves).
// Matches CDNA5 16-bit A/B WMMA VGPR layouts when (p,gap) chosen per-lane.
__device__ inline bf16x16 ld2(const u16* p, int gap) {
    union { bf16x16 v; u32x4 q[2]; } u;
    u.q[0] = *(const u32x4*)(p);
    u.q[1] = *(const u32x4*)(p + gap);
    return u.v;
}

__device__ inline f32x8 wmma_bf16(bf16x16 a, bf16x16 b, f32x8 c) {
    // D = A(16x32) * B(32x16) + C, f32 accumulate
    return __builtin_amdgcn_wmma_f32_16x16x32_bf16(
        /*neg_a=*/false, a, /*neg_b=*/false, b,
        /*c_mod=*/(short)0, c, /*reuse_a=*/false, /*reuse_b=*/false);
}

// ---------------------------------------------------------------------------
// Kernel 1: fp32 -> bf16 elementwise convert
// ---------------------------------------------------------------------------
__global__ void __launch_bounds__(256)
k_f32_to_bf16(const float* __restrict__ s, u16* __restrict__ d, int n) {
    int i = blockIdx.x * 256 + threadIdx.x;
    if (i < n) d[i] = f2bf(s[i]);
}

// ---------------------------------------------------------------------------
// Kernel 2: pack q_w [N,D,H] + kv_w [2,D,H] into bf16 B matrix [D][2560]
// cols 0..2047: q (n=col>>8, h=col&255) ; 2048..2303: k ; 2304..2559: v
// ---------------------------------------------------------------------------
__global__ void __launch_bounds__(256)
k_pack_w(const float* __restrict__ qw, const float* __restrict__ kvw,
         u16* __restrict__ dst, int D) {
    long i = (long)blockIdx.x * 256 + threadIdx.x;
    long total = (long)D * QKVC;
    if (i >= total) return;
    int d   = (int)(i / QKVC);
    int col = (int)(i % QKVC);
    float v;
    if (col < 2048) {
        int n = col >> 8, h = col & 255;
        v = qw[((size_t)n * D + d) * H_ + h];
    } else if (col < 2304) {
        v = kvw[(size_t)d * H_ + (col - 2048)];
    } else {
        v = kvw[(size_t)D * H_ + (size_t)d * H_ + (col - 2304)];
    }
    dst[i] = f2bf(v);
}

// ---------------------------------------------------------------------------
// Kernel 3: generic tiled bf16 GEMM, fp32 accumulate, double-buffered LDS
// with a register pipeline stage:
//   issue global loads(chunk i+1) -> WMMA(chunk i) -> LDS store(chunk i+1)
// so s_wait_loadcnt lands AFTER the matrix math.
//   C[M,Nc] = A[M,Ka](bf16) @ Bm[Ka,Nc](bf16)
// A-row remap: src_row = (row / rows_per_b) * batch_stride + row % rows_per_b
// (hoisted out of the K loop into per-thread staging pointers).
// WG: 256 thr = 8 waves, 128x128 tile; wave tile 64x32 (4x2 accum tiles).
// ---------------------------------------------------------------------------
#define GP 40   // LDS row pitch in halves (80B: 16B-aligned, conflict-spread)
__global__ void __launch_bounds__(256)
k_gemm_bf16(const u16* __restrict__ A, const u16* __restrict__ Bm,
            float* __restrict__ C, int M, int Ka, int Nc,
            int rows_per_b, int batch_stride) {
    __shared__ u16 Ash[2][128 * GP];   // [r][k] row-major, K-chunk = 32
    __shared__ u16 Bts[2][128 * GP];   // transposed: [col][k]

    const int tid  = threadIdx.x;
    const int w    = tid >> 5, lane = tid & 31;
    const int lm   = lane & 15, sel = lane >> 4;
    const int wm   = w >> 2, wn = w & 3;               // 2 x 4 wave grid
    const int mB   = blockIdx.y * 128, nB = blockIdx.x * 128;

    // Loop-invariant staging descriptors: each thread moves 2x16B of A
    // (contiguous) and 2x16B of B (scattered into the transposed tile).
    const u16* aSrc[2]; int aOff[2];
    const u16* bSrc[2]; int bK[2], bC[2];
    #pragma unroll
    for (int j = 0; j < 2; ++j) {
        unsigned idx = (unsigned)tid + 256u * j;       // 0..511
        unsigned r = idx >> 2, c8 = (idx & 3u) * 8u;   // A: 128 rows x 4 quads
        unsigned gr = (unsigned)mB + r;
        unsigned sr = (gr / (unsigned)rows_per_b) * (unsigned)batch_stride
                    + (gr % (unsigned)rows_per_b);
        aSrc[j] = A + (size_t)sr * Ka + c8;
        aOff[j] = (int)(r * GP + c8);
        unsigned k = idx >> 4, cc = (idx & 15u) * 8u;  // B: 32 k x 16 quads
        bSrc[j] = Bm + (size_t)k * Nc + nB + cc;
        bK[j] = (int)k; bC[j] = (int)cc;
    }

    u32x4 aReg[2], bReg[2];
    auto loadChunk = [&](int k0) {      // issue only; no wait until LDS store
        #pragma unroll
        for (int j = 0; j < 2; ++j) {
            aReg[j] = *(const u32x4*)(aSrc[j] + k0);
            bReg[j] = *(const u32x4*)(bSrc[j] + (size_t)k0 * Nc);
            __builtin_prefetch(aSrc[j] + k0 + 64, 0, 1);           // +2 chunks
            __builtin_prefetch(bSrc[j] + (size_t)(k0 + 64) * Nc, 0, 1);
        }
    };
    auto storeChunk = [&](int buf) {
        #pragma unroll
        for (int j = 0; j < 2; ++j) {
            *(u32x4*)&Ash[buf][aOff[j]] = aReg[j];
            union { u32x4 v; u16 h[8]; } bu; bu.v = bReg[j];
            #pragma unroll
            for (int e = 0; e < 8; ++e)
                Bts[buf][(bC[j] + e) * GP + bK[j]] = bu.h[e];
        }
    };

    f32x8 acc[4][2];
    #pragma unroll
    for (int mt = 0; mt < 4; ++mt)
        #pragma unroll
        for (int nt = 0; nt < 2; ++nt) acc[mt][nt] = fzero8();

    loadChunk(0);
    storeChunk(0);
    __syncthreads();
    const int nch = Ka >> 5;
    for (int i = 0; i < nch; ++i) {
        const int buf = i & 1;
        const bool more = (i + 1 < nch);
        if (more) loadChunk((i + 1) << 5);   // global loads fly during WMMA

        bf16x16 af[4];
        #pragma unroll
        for (int mt = 0; mt < 4; ++mt)
            af[mt] = ld2(&Ash[buf][(wm * 64 + mt * 16 + lm) * GP + sel * 8], 16);
        #pragma unroll
        for (int nt = 0; nt < 2; ++nt) {
            bf16x16 bf = ld2(&Bts[buf][(wn * 32 + nt * 16 + lm) * GP + sel * 8], 16);
            #pragma unroll
            for (int mt = 0; mt < 4; ++mt)
                acc[mt][nt] = wmma_bf16(af[mt], bf, acc[mt][nt]);
        }

        if (more) storeChunk(buf ^ 1);       // wait_loadcnt lands here
        __syncthreads();   // one barrier per chunk: read(buf) & write(buf^1) done
    }

    // epilogue (C layout: row = v + 8*sel, col = lane%16)
    #pragma unroll
    for (int mt = 0; mt < 4; ++mt)
        #pragma unroll
        for (int nt = 0; nt < 2; ++nt)
            #pragma unroll
            for (int v = 0; v < 8; ++v) {
                int row = mB + wm * 64 + mt * 16 + v + 8 * sel;
                int col = nB + wn * 32 + nt * 16 + lm;
                C[(size_t)row * Nc + col] = acc[mt][nt][v];
            }
}

// ---------------------------------------------------------------------------
// Kernel 4: RoPE + scale + scatter into attention layouts (bf16)
//   Qb [B,NH,T,H], Kb [B,T,H], Vb [B,T,H]
// one block per (b,t); q rows come from C0 (t<S0) or C1.
// ---------------------------------------------------------------------------
__global__ void __launch_bounds__(256)
k_rope(const float* __restrict__ C0, const float* __restrict__ C1,
       u16* __restrict__ Qb, u16* __restrict__ Kb, u16* __restrict__ Vb) {
    const int bt = blockIdx.x;
    const int b = bt / T_, t = bt % T_;
    const int tid = threadIdx.x;
    const float* row = (t < S0_)
        ? C0 + ((size_t)b * S0_ + t) * QKVC
        : C1 + ((size_t)b * S1_ + (t - S0_)) * QKVC;
    const float qscale = 0.0625f;   // H^-0.5

    // q heads: 8 heads x 128 rotation pairs
    for (int i = tid; i < NH_ * 128; i += 256) {
        int n = i >> 7, h = i & 127;
        float x1 = row[n * H_ + h], x2 = row[n * H_ + h + 128];
        float inv = __powf(10000.0f, -(float)h * (1.0f / 128.0f));
        float r = (float)t * inv, s = __sinf(r), c = __cosf(r);
        size_t base = (((size_t)b * NH_ + n) * T_ + t) * H_;
        Qb[base + h]       = f2bf((x1 * c - x2 * s) * qscale);
        Qb[base + h + 128] = f2bf((x2 * c + x1 * s) * qscale);
    }
    // k head
    if (tid < 128) {
        int h = tid;
        float x1 = row[2048 + h], x2 = row[2048 + h + 128];
        float inv = __powf(10000.0f, -(float)h * (1.0f / 128.0f));
        float r = (float)t * inv, s = __sinf(r), c = __cosf(r);
        size_t base = ((size_t)b * T_ + t) * H_;
        Kb[base + h]       = f2bf(x1 * c - x2 * s);
        Kb[base + h + 128] = f2bf(x2 * c + x1 * s);
    }
    // v head
    {
        int h = tid;
        Vb[((size_t)b * T_ + t) * H_ + h] = f2bf(row[2304 + h]);
    }
}

// ---------------------------------------------------------------------------
// Kernel 5: flash attention (online softmax), causal.
// grid: B*NH*(T/128) blocks, 256 thr = 8 waves, each wave = 16 query rows.
// Key tile = 32 keys. Enc out: [B,T,NH,H] bf16.
// ---------------------------------------------------------------------------
#define KROW 272   // 256 + 16 pad halves (row = 544B, 16B aligned)
#define VROW 40    // 32 + 8 pad halves (row = 80B)
#define PROW 40
__global__ void __launch_bounds__(256)
k_attn(const u16* __restrict__ Qb, const u16* __restrict__ Kb,
       const u16* __restrict__ Vb, u16* __restrict__ Enc) {
    __shared__ u16 Ksh[32 * KROW];        // [key][h]
    __shared__ u16 Vtsh[H_ * VROW];       // transposed: [h][key]
    __shared__ u16 Psh[8 * 16 * PROW];    // per-wave P tile 16x32 bf16

    const int bx = blockIdx.x;
    const int qt = bx & 7, head = (bx >> 3) & 7, b = bx >> 6;
    const int tid = threadIdx.x;
    const int w = tid >> 5, lane = tid & 31, lm = lane & 15, sel = lane >> 4;
    const int qbase = qt * 128 + w * 16;          // this wave's 16 q rows

    // Q fragments resident in registers (16 rows x 256 halves)
    const u16* qp = Qb + (((size_t)b * NH_ + head) * T_ + qbase + lm) * H_;
    bf16x16 qf[8];
    #pragma unroll
    for (int hc = 0; hc < 8; ++hc)
        qf[hc] = ld2(qp + hc * 32 + sel * 8, 16);

    f32x8 o[16];
    #pragma unroll
    for (int nt = 0; nt < 16; ++nt) o[nt] = fzero8();
    float mreg[8], lreg[8];
    #pragma unroll
    for (int v = 0; v < 8; ++v) { mreg[v] = -3.0e38f; lreg[v] = 0.0f; }

    // per-thread staging coordinates (loop-invariant)
    const int skey = tid >> 5;                 // base key row handled
    const int sh8  = (tid & 31) * 8;           // 16B column quad
    const u16* kBase = Kb + (size_t)b * T_ * H_;
    const u16* vBase = Vb + (size_t)b * T_ * H_;

    const int kend = qt * 128 + 128;   // causal upper bound for this WG
    for (int kb = 0; kb < kend; kb += 32) {
        __syncthreads();
        // stage K tile [32][256] and V tile transposed [256][32], 16B loads
        #pragma unroll
        for (int j = 0; j < 4; ++j) {
            int key = skey + j * 8;
            *(u32x4*)&Ksh[key * KROW + sh8] =
                *(const u32x4*)&kBase[(size_t)(kb + key) * H_ + sh8];
            union { u32x4 v; u16 h[8]; } tv;
            tv.v = *(const u32x4*)&vBase[(size_t)(kb + key) * H_ + sh8];
            #pragma unroll
            for (int e = 0; e < 8; ++e)
                Vtsh[(sh8 + e) * VROW + key] = tv.h[e];
        }
        __syncthreads();

        // S = Q K^T : two 16x16 tiles over 8 H-chunks
        f32x8 s0 = fzero8(), s1 = fzero8();
        #pragma unroll
        for (int hc = 0; hc < 8; ++hc) {
            bf16x16 kf0 = ld2(&Ksh[(lm) * KROW + hc * 32 + sel * 8], 16);
            bf16x16 kf1 = ld2(&Ksh[(16 + lm) * KROW + hc * 32 + sel * 8], 16);
            s0 = wmma_bf16(qf[hc], kf0, s0);
            s1 = wmma_bf16(qf[hc], kf1, s1);
        }

        // online softmax: row stats shared by 16-lane groups (C-tile rows)
        float scv[8];
        #pragma unroll
        for (int v = 0; v < 8; ++v) {
            int qr = qbase + v + 8 * sel;
            float e0 = (kb + lm      <= qr) ? s0[v] : BIG_NEG;
            float e1 = (kb + 16 + lm <= qr) ? s1[v] : BIG_NEG;
            float tmx = fmaxf(e0, e1);
            #pragma unroll
            for (int msk = 1; msk < 16; msk <<= 1)
                tmx = fmaxf(tmx, __shfl_xor(tmx, msk, 32));
            float mn = fmaxf(mreg[v], tmx);
            float sc = __expf(mreg[v] - mn);
            float p0 = __expf(e0 - mn);
            float p1 = __expf(e1 - mn);
            Psh[w * (16 * PROW) + (v + 8 * sel) * PROW + lm]      = f2bf(p0);
            Psh[w * (16 * PROW) + (v + 8 * sel) * PROW + 16 + lm] = f2bf(p1);
            float rs = p0 + p1;
            #pragma unroll
            for (int msk = 1; msk < 16; msk <<= 1)
                rs += __shfl_xor(rs, msk, 32);
            lreg[v] = lreg[v] * sc + rs;
            mreg[v] = mn;
            scv[v]  = sc;
        }

        // rescale running O, then O += P @ V
        #pragma unroll
        for (int nt = 0; nt < 16; ++nt)
            #pragma unroll
            for (int v = 0; v < 8; ++v) o[nt][v] *= scv[v];

        bf16x16 pA = ld2(&Psh[w * (16 * PROW) + lm * PROW + sel * 8], 16);
        #pragma unroll
        for (int nt = 0; nt < 16; ++nt) {
            bf16x16 vB = ld2(&Vtsh[(nt * 16 + lm) * VROW + sel * 8], 16);
            o[nt] = wmma_bf16(pA, vB, o[nt]);
        }
    }

    // normalize and write encoded [B,T,NH,H] bf16
    float inv[8];
    #pragma unroll
    for (int v = 0; v < 8; ++v) inv[v] = 1.0f / lreg[v];
    #pragma unroll
    for (int nt = 0; nt < 16; ++nt)
        #pragma unroll
        for (int v = 0; v < 8; ++v) {
            int t = qbase + v + 8 * sel;
            Enc[(((size_t)b * T_ + t) * NH_ + head) * H_ + nt * 16 + lm] =
                f2bf(o[nt][v] * inv[v]);
        }
}

// ---------------------------------------------------------------------------
// host launcher
// ---------------------------------------------------------------------------
extern "C" void kernel_launch(void* const* d_in, const int* in_sizes, int n_in,
                              void* d_out, int out_size, void* d_ws, size_t ws_size,
                              hipStream_t stream) {
    const float* x0    = (const float*)d_in[0];
    const float* x1    = (const float*)d_in[1];
    const float* q0_w  = (const float*)d_in[4];
    const float* kv0_w = (const float*)d_in[5];
    const float* q1_w  = (const float*)d_in[6];
    const float* kv1_w = (const float*)d_in[7];
    const float* o0_w  = (const float*)d_in[8];
    const float* o1_w  = (const float*)d_in[9];
    float* out = (float*)d_out;

    // carve workspace
    char* p = (char*)d_ws;
    auto take = [&](size_t bytes) { char* r = p; p += (bytes + 255) & ~size_t(255); return r; };
    u16*   xb0 = (u16*)  take((size_t)B_ * S0_ * D0_ * 2);
    u16*   xb1 = (u16*)  take((size_t)B_ * S1_ * D1_ * 2);
    u16*   W0  = (u16*)  take((size_t)D0_ * QKVC * 2);
    u16*   W1  = (u16*)  take((size_t)D1_ * QKVC * 2);
    u16*   Wo0 = (u16*)  take((size_t)NH_ * H_ * D0_ * 2);
    u16*   Wo1 = (u16*)  take((size_t)NH_ * H_ * D1_ * 2);
    float* C0  = (float*)take((size_t)B_ * S0_ * QKVC * 4);
    float* C1  = (float*)take((size_t)B_ * S1_ * QKVC * 4);
    u16*   Qb  = (u16*)  take((size_t)B_ * NH_ * T_ * H_ * 2);
    u16*   Kb  = (u16*)  take((size_t)B_ * T_ * H_ * 2);
    u16*   Vb  = (u16*)  take((size_t)B_ * T_ * H_ * 2);
    u16*   Enc = (u16*)  take((size_t)B_ * T_ * NH_ * H_ * 2);

    // 1. bf16 conversions
    {
        int n0 = B_ * S0_ * D0_;
        k_f32_to_bf16<<<(n0 + 255) / 256, 256, 0, stream>>>(x0, xb0, n0);
        int n1 = B_ * S1_ * D1_;
        k_f32_to_bf16<<<(n1 + 255) / 256, 256, 0, stream>>>(x1, xb1, n1);
        int no0 = NH_ * H_ * D0_;
        k_f32_to_bf16<<<(no0 + 255) / 256, 256, 0, stream>>>(o0_w, Wo0, no0);
        int no1 = NH_ * H_ * D1_;
        k_f32_to_bf16<<<(no1 + 255) / 256, 256, 0, stream>>>(o1_w, Wo1, no1);
    }
    // 2. fused QKV weight packs
    k_pack_w<<<(D0_ * QKVC + 255) / 256, 256, 0, stream>>>(q0_w, kv0_w, W0, D0_);
    k_pack_w<<<(D1_ * QKVC + 255) / 256, 256, 0, stream>>>(q1_w, kv1_w, W1, D1_);

    // 3. QKV projection GEMMs (identity row map)
    k_gemm_bf16<<<dim3(QKVC / 128, (B_ * S0_) / 128), 256, 0, stream>>>(
        xb0, W0, C0, B_ * S0_, D0_, QKVC, B_ * S0_, 0);
    k_gemm_bf16<<<dim3(QKVC / 128, (B_ * S1_) / 128), 256, 0, stream>>>(
        xb1, W1, C1, B_ * S1_, D1_, QKVC, B_ * S1_, 0);

    // 4. RoPE + scatter
    k_rope<<<B_ * T_, 256, 0, stream>>>(C0, C1, Qb, Kb, Vb);

    // 5. flash attention
    k_attn<<<B_ * NH_ * (T_ / 128), 256, 0, stream>>>(Qb, Kb, Vb, Enc);

    // 6. output projections straight into d_out
    //    out0: rows map (b,t) -> Enc row b*T + t           (t < S0)
    k_gemm_bf16<<<dim3(D0_ / 128, (B_ * S0_) / 128), 256, 0, stream>>>(
        Enc, Wo0, out, B_ * S0_, NH_ * H_, D0_, S0_, T_);
    //    out1: Enc pointer offset by S0 rows; rows map (b,t) -> b*T + t
    k_gemm_bf16<<<dim3(D1_ / 128, (B_ * S1_) / 128), 256, 0, stream>>>(
        Enc + (size_t)S0_ * NH_ * H_, Wo1, out + (size_t)B_ * S0_ * D0_,
        B_ * S1_, NH_ * H_, D1_, S1_, T_);
}
